// SparseMoe_34050500723053
// MI455X (gfx1250) — matrix operand order repaired and checked by reading, not compiled, not run on hardware
//
#include <hip/hip_runtime.h>
#include <stdint.h>

// ---------------- problem constants (from reference setup_inputs) ----------
#define T_TOK 16384   // B*S = 4*4096
#define D_    1024
#define H_    4096
#define E_    8
#define CAP   16384   // worst-case tokens per expert
#define MT    64      // token rows per workgroup tile
#define HC    512     // hidden chunk kept in LDS
#define XS_STRIDE 1032 // bf16 elems per x row in LDS (pad 8 -> 2064B, 16B aligned, bank-skewed)
#define HS_STRIDE 520  // bf16 elems per h row in LDS (pad 8 -> 1040B, 16B aligned, bank-skewed)

typedef __attribute__((ext_vector_type(16))) __bf16 bf16x16;
typedef __attribute__((ext_vector_type(8)))  float  f32x8;

union Frag { bf16x16 v; uint4 q[2]; };

__device__ __forceinline__ unsigned short f2bf(float f) {
  union { float f; uint32_t u; } c; c.f = f;
  uint32_t u = c.u;
  u += 0x7FFFu + ((u >> 16) & 1u);   // round-to-nearest-even
  return (unsigned short)(u >> 16);
}

// ---------------- router: logits, top-2, softmax, per-expert compaction ----
__global__ void moe_router(const float* __restrict__ x, const float* __restrict__ Wg,
                           const float* __restrict__ bg, int* __restrict__ counts,
                           int* __restrict__ toks, float* __restrict__ gates) {
  int t = blockIdx.x * blockDim.x + threadIdx.x;
  if (t >= T_TOK) return;
  float lg[E_];
#pragma unroll
  for (int e = 0; e < E_; ++e) lg[e] = bg[e];
  const float* xr = x + (size_t)t * D_;
  for (int k = 0; k < D_; ++k) {
    float xv = xr[k];
#pragma unroll
    for (int e = 0; e < E_; ++e) lg[e] += xv * Wg[k * E_ + e];   // Wg is (D,E)
  }
  // top-2 with lax.top_k tie semantics (first/lowest index wins)
  int i0 = 0; float v0 = lg[0];
#pragma unroll
  for (int e = 1; e < E_; ++e) if (lg[e] > v0) { v0 = lg[e]; i0 = e; }
  int i1 = -1; float v1 = -3.4e38f;
#pragma unroll
  for (int e = 0; e < E_; ++e) if (e != i0 && lg[e] > v1) { v1 = lg[e]; i1 = e; }
  float ex  = __expf(v1 - v0);
  float inv = 1.0f / (1.0f + ex);
  float p0 = inv, p1 = ex * inv;     // softmax over the two kept logits
  int p = atomicAdd(&counts[i0], 1);
  toks[i0 * CAP + p] = t; gates[i0 * CAP + p] = p0;
  p = atomicAdd(&counts[i1], 1);
  toks[i1 * CAP + p] = t; gates[i1 * CAP + p] = p1;
}

// ---------------- f32 -> bf16 weight conversion into WMMA B-fragment order -
// Output: 1KB blocks indexed by ((e*NT + nt)*KC + kc); inside a block lane L
// owns 32 contiguous bytes = 8 packed dwords. B(32x16 bf16) ISA layout:
// lanes 0-15 hold K=0..7,16..23 (2/VGPR), lanes 16-31 hold K=8..15,24..31; lane%16 = N.
__global__ void conv_swz(const float* __restrict__ W, uint32_t* __restrict__ outb,
                         int K, int N, int KC, int NT) {
  int blk = blockIdx.x;                 // (e*NT + nt)*KC + kc
  int kc  = blk % KC;
  int ent = blk / KC;
  int nt  = ent % NT;
  int e   = ent / NT;
  int tid = threadIdx.x;                // 256 = 32 lanes * 8 dwords
  int d = tid & 7, l = tid >> 3;
  int n  = nt * 16 + (l & 15);
  int kb = kc * 32 + ((l < 16) ? 0 : 8) + d * 2 + ((d < 4) ? 0 : 8);
  const float* base = W + (size_t)e * K * N;       // W[e][k][n]
  unsigned short lo = f2bf(base[(size_t)kb * N + n]);
  unsigned short hi = f2bf(base[(size_t)(kb + 1) * N + n]);
  outb[(size_t)blk * 256 + tid] = (uint32_t)lo | ((uint32_t)hi << 16);
}

// ---------------- fused expert FFN: relu(x@W1+b1)@W2+b2, gated scatter -----
__global__ __launch_bounds__(512) void moe_ffn(
    const float* __restrict__ x, const float* __restrict__ b1,
    const float* __restrict__ b2, const int* __restrict__ counts,
    const int* __restrict__ toks, const float* __restrict__ gates,
    const uint32_t* __restrict__ w1b, const uint32_t* __restrict__ w2b,
    float* __restrict__ out) {
  int e = blockIdx.x >> 8;          // 256 max tiles per expert
  int tile = blockIdx.x & 255;
  int cnt = counts[e];
  if (tile * MT >= cnt) return;     // fixed grid, dynamic routing -> early exit
  int nrows = cnt - tile * MT; if (nrows > MT) nrows = MT;

  extern __shared__ char smem[];
  unsigned short* xs = (unsigned short*)smem;                             // MT*XS_STRIDE bf16
  unsigned short* hs = (unsigned short*)(smem + MT * XS_STRIDE * 2);      // MT*HS_STRIDE bf16
  int*   tok_s  = (int*)(smem + MT * XS_STRIDE * 2 + MT * HS_STRIDE * 2);
  float* gate_s = (float*)(tok_s + MT);

  int tid = threadIdx.x;
  if (tid < MT) {
    int tk = -1; float g = 0.f;
    if (tid < nrows) {
      tk = toks [e * CAP + tile * MT + tid];
      g  = gates[e * CAP + tile * MT + tid];
    }
    tok_s[tid] = tk; gate_s[tid] = g;
  }
  __syncthreads();
  // gather 64 token rows, convert f32 -> bf16 into LDS (coalesced along row)
  for (int i = tid; i < MT * D_; i += 512) {
    int r = i >> 10, c = i & (D_ - 1);
    int tk = tok_s[r];
    float v = (tk >= 0) ? x[(size_t)tk * D_ + c] : 0.f;
    xs[r * XS_STRIDE + c] = f2bf(v);
  }
  __syncthreads();

  const int w  = tid >> 5;          // wave 0..15
  const int l  = tid & 31;
  const int lh = (l < 16) ? 0 : 1;  // lane half (K split for A, M split for C/D)
  const int lm = l & 15;

  f32x8 zero = {};
  f32x8 acc[4][4];                  // phase-2 accumulators: 4 M-tiles x 4 N-tiles (w*64 cols)
#pragma unroll
  for (int m = 0; m < 4; ++m)
#pragma unroll
    for (int n = 0; n < 4; ++n) acc[m][n] = zero;

  // per-lane LDS A-fragment bases (row + lane-half K offset folded in once)
  const unsigned short* axs[4];
  const unsigned short* ahs[4];
#pragma unroll
  for (int mt = 0; mt < 4; ++mt) {
    axs[mt] = xs + (mt * 16 + lm) * XS_STRIDE + lh * 8;
    ahs[mt] = hs + (mt * 16 + lm) * HS_STRIDE + lh * 8;
  }

  for (int hc = 0; hc < H_ / HC; ++hc) {
    // -------- phase 1: hs(64 x 512) = relu(xs @ W1[:, hc*512 ..] + b1) -----
    f32x8 acc1[4][2];               // wave covers 32 cols of the chunk
#pragma unroll
    for (int m = 0; m < 4; ++m)
#pragma unroll
      for (int j = 0; j < 2; ++j) acc1[m][j] = zero;

    // per-lane weight base pointers: tile = 32 kc-blocks * 64 uint4
    const uint4* bb1[2];
#pragma unroll
    for (int j = 0; j < 2; ++j)
      bb1[j] = (const uint4*)w1b +
               (size_t)(e * 256 + hc * 32 + w * 2 + j) * 2048 + l * 2;

#pragma unroll 2
    for (int kc = 0; kc < 32; ++kc) {       // K = 1024 in steps of 32
      Frag b[2], a[4];
      // long-latency global loads first; address = base + kc*1024B (imm-foldable)
#pragma unroll
      for (int j = 0; j < 2; ++j) { b[j].q[0] = bb1[j][kc * 64]; b[j].q[1] = bb1[j][kc * 64 + 1]; }
      // branchless near-cache prefetch 8 K-steps ahead (wraps inside the tile)
      __builtin_prefetch(bb1[0] + ((kc + 8) & 31) * 64, 0, 3);
#pragma unroll
      for (int mt = 0; mt < 4; ++mt) {
        a[mt].q[0] = *(const uint4*)(axs[mt] + kc * 32);        // ds_load_b128
        a[mt].q[1] = *(const uint4*)(axs[mt] + kc * 32 + 16);
      }
#pragma unroll
      for (int j = 0; j < 2; ++j)
#pragma unroll
        for (int mt = 0; mt < 4; ++mt)
          acc1[mt][j] = __builtin_amdgcn_wmma_f32_16x16x32_bf16(
              false, a[mt].v, false, b[j].v, (short)0, acc1[mt][j], false, false);
    }
    // bias + relu + bf16 store of h chunk into LDS
#pragma unroll
    for (int j = 0; j < 2; ++j) {
      int cl = w * 32 + j * 16 + lm;        // column within chunk
      float b1v = b1[e * H_ + hc * HC + cl];
#pragma unroll
      for (int mt = 0; mt < 4; ++mt) {
#pragma unroll
        for (int r = 0; r < 8; ++r) {       // C/D layout: M = r + lh*8
          float v = acc1[mt][j][r] + b1v;
          v = fmaxf(v, 0.f);
          hs[(mt * 16 + lh * 8 + r) * HS_STRIDE + cl] = f2bf(v);
        }
      }
    }
    __syncthreads();
    // -------- phase 2: acc += hs @ W2[hc*512 .. , :] -----------------------
    // per-lane weight bases: tile = 128 kc-blocks * 64 uint4; start at kcg = hc*16
    const uint4* bb2[4];
#pragma unroll
    for (int n = 0; n < 4; ++n)
      bb2[n] = (const uint4*)w2b +
               (size_t)(e * 64 + w * 4 + n) * 8192 + (size_t)(hc * 16) * 64 + l * 2;

#pragma unroll 2
    for (int kc = 0; kc < 16; ++kc) {       // HC = 512 in steps of 32
      Frag b[4], a[4];
#pragma unroll
      for (int n = 0; n < 4; ++n) { b[n].q[0] = bb2[n][kc * 64]; b[n].q[1] = bb2[n][kc * 64 + 1]; }
      __builtin_prefetch(bb2[0] + ((kc + 8) & 15) * 64, 0, 3);
#pragma unroll
      for (int mt = 0; mt < 4; ++mt) {
        a[mt].q[0] = *(const uint4*)(ahs[mt] + kc * 32);
        a[mt].q[1] = *(const uint4*)(ahs[mt] + kc * 32 + 16);
      }
#pragma unroll
      for (int n = 0; n < 4; ++n)
#pragma unroll
        for (int mt = 0; mt < 4; ++mt)
          acc[mt][n] = __builtin_amdgcn_wmma_f32_16x16x32_bf16(
              false, a[mt].v, false, b[n].v, (short)0, acc[mt][n], false, false);
    }
    __syncthreads();                        // hs reused by next chunk's phase 1
  }
  // -------- epilogue: out[tok] += gate * (acc + b2) (exactly 2 adds/elem) --
#pragma unroll
  for (int n = 0; n < 4; ++n) {
    int col = w * 64 + n * 16 + lm;
    float b2v = b2[e * D_ + col];
#pragma unroll
    for (int mt = 0; mt < 4; ++mt) {
#pragma unroll
      for (int r = 0; r < 8; ++r) {
        int row = mt * 16 + lh * 8 + r;
        int tk = tok_s[row];
        if (tk >= 0)
          atomicAdd(out + (size_t)tk * D_ + col,
                    gate_s[row] * (acc[mt][n][r] + b2v));
      }
    }
  }
}

// ---------------- host launcher -------------------------------------------
extern "C" void kernel_launch(void* const* d_in, const int* in_sizes, int n_in,
                              void* d_out, int out_size, void* d_ws, size_t ws_size,
                              hipStream_t stream) {
  const float* x  = (const float*)d_in[0];
  const float* Wg = (const float*)d_in[1];
  const float* bg = (const float*)d_in[2];
  const float* W1 = (const float*)d_in[3];
  const float* b1 = (const float*)d_in[4];
  const float* W2 = (const float*)d_in[5];
  const float* b2 = (const float*)d_in[6];
  // d_in[7] = topk_num (== 2, hardcoded in the kernels)
  float* out = (float*)d_out;

  // workspace layout (needs ~130 MB): counts | tok lists | gate lists | bf16 W1 | bf16 W2
  char* ws = (char*)d_ws;
  int*      counts = (int*)ws;                                  // 32 B
  int*      toks   = (int*)(ws + 256);                          // 8*16384*4 = 512 KB
  float*    gatesp = (float*)(ws + 256 + (size_t)E_ * CAP * 4); // 512 KB
  uint32_t* w1b    = (uint32_t*)(ws + ((size_t)2  << 20));      // 64 MB
  uint32_t* w2b    = (uint32_t*)(ws + ((size_t)66 << 20));      // 64 MB
  if (ws_size < ((size_t)130 << 20)) return;                    // cannot run without scratch

  hipMemsetAsync(d_out, 0, (size_t)out_size * sizeof(float), stream);
  hipMemsetAsync(counts, 0, E_ * sizeof(int), stream);

  // W1: (E,1024,4096) -> NT=256 N-tiles, KC=32 K-chunks ; W2: (E,4096,1024) -> NT=64, KC=128
  conv_swz<<<E_ * 256 * 32,  256, 0, stream>>>(W1, w1b, D_, H_, 32, 256);
  conv_swz<<<E_ * 64  * 128, 256, 0, stream>>>(W2, w2b, H_, D_, 128, 64);
  moe_router<<<T_TOK / 256, 256, 0, stream>>>(x, Wg, bg, counts, toks, gatesp);

  const int SMEM = MT * XS_STRIDE * 2 + MT * HS_STRIDE * 2 + MT * 8; // 199168 B < 320 KB WGP LDS
  (void)hipFuncSetAttribute((const void*)moe_ffn,
                            hipFuncAttributeMaxDynamicSharedMemorySize, SMEM);
  moe_ffn<<<E_ * 256, 512, SMEM, stream>>>(x, b1, b2, counts, toks, gatesp,
                                           w1b, w2b, out);
  (void)in_sizes; (void)n_in;
}
